// JambaSparseMoeBlock_47287589929501
// MI455X (gfx1250) — matrix operand order repaired and checked by reading, not compiled, not run on hardware
//
#include <hip/hip_runtime.h>
#include <hip/hip_bf16.h>

#define HIDDEN 1024
#define INTER  4096
#define N_EXPERTS 8
#define M_TILE 64            // tokens per block (4 WMMA row-tiles)
#define ROW_TILES 4          // M_TILE / 16
#define NQ 4                 // INTER split across blocks
#define H_CHUNK (INTER / NQ) // 1024 h-columns per block
#define WAVES  8             // 256 threads / wave32

typedef __attribute__((ext_vector_type(16))) __bf16 bf16x16;
typedef __attribute__((ext_vector_type(8)))  __bf16 bf16x8;
typedef __attribute__((ext_vector_type(2)))  __bf16 bf16x2;
typedef __attribute__((ext_vector_type(8)))  float  v8f;

// fp32 pair -> packed bf16 dword: round-half-up + byte-select (v_perm_b32)
__device__ __forceinline__ unsigned pack2bf(float e0, float e1) {
    union { float f; unsigned u; } a, b; a.f = e0; b.f = e1;
    const unsigned r0 = a.u + 0x8000u;
    const unsigned r1 = b.u + 0x8000u;
    // sel 0x07060302: D = {S0[31:24],S0[23:16],S1[31:24],S1[23:16]}
    return __builtin_amdgcn_perm(r1, r0, 0x07060302u);
}

__device__ __forceinline__ __bf16 f2bf(float f) {
    union { float f; unsigned u; } in; in.f = f;
    const unsigned u = in.u + 0x8000u;
    union { unsigned short s; __bf16 b; } out; out.s = (unsigned short)(u >> 16);
    return out.b;
}

// ---------------------------------------------------------------- init ----
__global__ void zero_counts_kernel(int* counts) {
    if (threadIdx.x < N_EXPERTS) counts[threadIdx.x] = 0;
}

// -------------------------------------------------------------- router ----
__global__ void router_kernel(const float* __restrict__ x,
                              const float* __restrict__ router_w,
                              float* __restrict__ logits_out,   // [T, E]
                              float* __restrict__ combine,      // [T, E]
                              int*   __restrict__ counts,       // [E]
                              int*   __restrict__ lists,        // [E, T] entries (t<<1)|slot
                              int T) {
    const int wave = threadIdx.x >> 5;
    const int lane = threadIdx.x & 31;
    const int t = blockIdx.x * (blockDim.x >> 5) + wave;
    if (t >= T) return;

    const float* xt = x + (size_t)t * HIDDEN;
    float xr[HIDDEN / 32];
    #pragma unroll
    for (int i = 0; i < HIDDEN / 32; ++i) xr[i] = xt[lane + 32 * i];

    float lg[N_EXPERTS];
    #pragma unroll
    for (int e = 0; e < N_EXPERTS; ++e) {
        const float* we = router_w + (size_t)e * HIDDEN;
        float s = 0.f;
        #pragma unroll
        for (int i = 0; i < HIDDEN / 32; ++i) s += xr[i] * we[lane + 32 * i];
        #pragma unroll
        for (int off = 16; off > 0; off >>= 1) s += __shfl_xor(s, off, 32);
        lg[e] = s;
    }

    if (lane == 0) {
        float mx = lg[0];
        #pragma unroll
        for (int e = 1; e < N_EXPERTS; ++e) mx = fmaxf(mx, lg[e]);
        float p[N_EXPERTS], den = 0.f;
        #pragma unroll
        for (int e = 0; e < N_EXPERTS; ++e) { p[e] = __expf(lg[e] - mx); den += p[e]; }
        const float inv = __builtin_amdgcn_rcpf(den);
        int i1 = 0;
        #pragma unroll
        for (int e = 1; e < N_EXPERTS; ++e) if (p[e] > p[i1]) i1 = e;
        int i2 = (i1 == 0) ? 1 : 0;
        #pragma unroll
        for (int e = 0; e < N_EXPERTS; ++e) if (e != i1 && p[e] > p[i2]) i2 = e;

        #pragma unroll
        for (int e = 0; e < N_EXPERTS; ++e) {
            logits_out[(size_t)t * N_EXPERTS + e] = lg[e];
            combine[(size_t)t * N_EXPERTS + e] = 0.f;
        }
        combine[(size_t)t * N_EXPERTS + i1] = p[i1] * inv;
        combine[(size_t)t * N_EXPERTS + i2] = p[i2] * inv;

        int pos1 = atomicAdd(&counts[i1], 1);
        lists[(size_t)i1 * T + pos1] = (t << 1) | 0;
        int pos2 = atomicAdd(&counts[i2], 1);
        lists[(size_t)i2 * T + pos2] = (t << 1) | 1;
    }
}

// ----------------------------------------------------- fused expert MLP ----
// grid = (ceil(T/64), NQ, E). Block owns 64 gathered tokens of one expert and
// one 1024-wide quarter of INTER: gate/up (K=1024) -> SwiGLU -> h quarter in
// LDS (bf16) -> partial down projection (K=1024) -> per-(token,slot,quarter)
// partial rows. One B fragment feeds 4 row-tile WMMAs (4x weight reuse).
__global__ __launch_bounds__(256, 1)
void moe_mlp_kernel(const float* __restrict__ x,
                    const float* __restrict__ gate_w,   // [E, INTER, HIDDEN]
                    const float* __restrict__ up_w,     // [E, INTER, HIDDEN]
                    const float* __restrict__ down_w,   // [E, HIDDEN, INTER]
                    const float* __restrict__ combine,  // [T, E]
                    const int*   __restrict__ counts,
                    const int*   __restrict__ lists,    // [E, T]
                    float*       __restrict__ contribq, // [T*2, NQ, HIDDEN]
                    int T) {
    const int e   = blockIdx.z;
    const int q   = blockIdx.y;
    const int cnt = counts[e];
    const int t0  = blockIdx.x * M_TILE;
    if (t0 >= cnt) return;

    const int tid  = threadIdx.x;
    const int wave = tid >> 5;
    const int lane = tid & 31;
    const int hs   = (lane < 16) ? 0 : 1;   // lane half select
    const int row  = lane & 15;             // row (A) / column (B/C) index
    const int kA   = hs * 8;                // unused (loadA also uses kB)
    const int kB   = hs * 16;               // A/B-frag K base (contiguous k = 16*hs + s)

    extern __shared__ char smem[];
    __bf16* xB = (__bf16*)smem;                                   // [64][HIDDEN]  128 KB
    __bf16* hB = (__bf16*)(smem + (size_t)M_TILE * HIDDEN * 2);   // [64][H_CHUNK] 128 KB

    __shared__ int   s_tok[M_TILE];
    __shared__ int   s_slot[M_TILE];
    __shared__ float s_w[M_TILE];
    __shared__ int   s_nvalid;

    if (tid < M_TILE) {
        int tok = 0, slot = 0; float w = 0.f;
        if (t0 + tid < cnt) {
            int entry = lists[(size_t)e * T + t0 + tid];
            tok = entry >> 1; slot = entry & 1;
            w = combine[(size_t)tok * N_EXPERTS + e];
        }
        s_tok[tid] = tok; s_slot[tid] = slot; s_w[tid] = w;
    }
    if (tid == 0) s_nvalid = (cnt - t0 < M_TILE) ? (cnt - t0) : M_TILE;
    __syncthreads();

    // gather x tile -> bf16 LDS (packed dword stores)
    for (int i = tid; i < M_TILE * HIDDEN / 4; i += 256) {
        const int m  = i / (HIDDEN / 4);
        const int dq = i % (HIDDEN / 4);
        const float4 v = ((const float4*)(x + (size_t)s_tok[m] * HIDDEN))[dq];
        unsigned* p = (unsigned*)(xB + (size_t)m * HIDDEN + dq * 4);
        p[0] = pack2bf(v.x, v.y);
        p[1] = pack2bf(v.z, v.w);
    }
    __syncthreads();

    // A fragment (16x32 bf16) from LDS row-major [.][stride], rows m0..m0+15
    auto loadA = [&](const __bf16* base, int stride, int m0, int k0) -> bf16x16 {
        union { bf16x16 v; bf16x8 h[2]; } u;
        const __bf16* p = base + (size_t)(m0 + row) * stride + k0 + kB;
        u.h[0] = *(const bf16x8*)(p);        // K = k0+kB .. +7
        u.h[1] = *(const bf16x8*)(p + 8);    // K = k0+kB+8 .. +15
        return u.v;
    };
    // B fragment (32x16 bf16): lane reads 16 contiguous fp32 of one weight row
    auto loadB = [&](const float* base, int ldk, int n0, int k0) -> bf16x16 {
        const float* p = base + (size_t)(n0 + row) * ldk + k0 + kB;
        const float4 a = ((const float4*)p)[0];
        const float4 b = ((const float4*)p)[1];
        const float4 c = ((const float4*)p)[2];
        const float4 d = ((const float4*)p)[3];
        union { bf16x16 v; unsigned u[8]; } r;
        r.u[0] = pack2bf(a.x, a.y); r.u[1] = pack2bf(a.z, a.w);
        r.u[2] = pack2bf(b.x, b.y); r.u[3] = pack2bf(b.z, b.w);
        r.u[4] = pack2bf(c.x, c.y); r.u[5] = pack2bf(c.z, c.w);
        r.u[6] = pack2bf(d.x, d.y); r.u[7] = pack2bf(d.z, d.w);
        return r.v;
    };

    // ---- phase 1: gate & up for this INTER quarter + SwiGLU -> hB ----
    const float* gbase = gate_w + (size_t)e * INTER * HIDDEN;
    const float* ubase = up_w   + (size_t)e * INTER * HIDDEN;
    const int hQ0 = q * H_CHUNK;
    for (int nt = wave; nt < H_CHUNK / 16; nt += WAVES) {
        const int h0  = hQ0 + nt * 16;   // global h (weight row)
        const int hl0 = nt * 16;         // local h (LDS)
        v8f aG[ROW_TILES] = {};
        v8f aU[ROW_TILES] = {};
        for (int dk = 0; dk < HIDDEN; dk += 32) {
            bf16x16 bg = loadB(gbase, HIDDEN, h0, dk);
            bf16x16 bu = loadB(ubase, HIDDEN, h0, dk);
            #pragma unroll
            for (int rt = 0; rt < ROW_TILES; ++rt) {     // 4x reuse of bg/bu
                bf16x16 a = loadA(xB, HIDDEN, rt * 16, dk);
                aG[rt] = __builtin_amdgcn_wmma_f32_16x16x32_bf16(false, a, false, bg,
                                                                 (short)0, aG[rt], false, false);
                aU[rt] = __builtin_amdgcn_wmma_f32_16x16x32_bf16(false, a, false, bu,
                                                                 (short)0, aU[rt], false, false);
            }
        }
        #pragma unroll
        for (int rt = 0; rt < ROW_TILES; ++rt) {
            #pragma unroll
            for (int v = 0; v < 8; ++v) {   // C layout: row = rt*16 + v + hs*8
                const float g = aG[rt][v];
                const float sig = __builtin_amdgcn_rcpf(1.f + __expf(-g));
                const float hval = g * sig * aU[rt][v];
                const int m = rt * 16 + v + hs * 8;
                hB[(size_t)m * H_CHUNK + hl0 + row] = f2bf(hval);
            }
        }
    }
    __syncthreads();

    // ---- phase 2: partial down projection over this quarter ----
    const float* dbase  = down_w + (size_t)e * HIDDEN * INTER;
    const int    nvalid = s_nvalid;
    for (int nt = wave; nt < HIDDEN / 16; nt += WAVES) {
        const int d0 = nt * 16;
        v8f acc[ROW_TILES] = {};
        for (int hk = 0; hk < H_CHUNK; hk += 32) {
            __builtin_prefetch(dbase + (size_t)(d0 + row) * INTER + hQ0 + hk + kB + 64, 0, 1);
            bf16x16 b = loadB(dbase, INTER, d0, hQ0 + hk);
            #pragma unroll
            for (int rt = 0; rt < ROW_TILES; ++rt) {     // 4x reuse of b
                bf16x16 a = loadA(hB, H_CHUNK, rt * 16, hk);
                acc[rt] = __builtin_amdgcn_wmma_f32_16x16x32_bf16(false, a, false, b,
                                                                  (short)0, acc[rt], false, false);
            }
        }
        #pragma unroll
        for (int rt = 0; rt < ROW_TILES; ++rt) {
            #pragma unroll
            for (int v = 0; v < 8; ++v) {
                const int m = rt * 16 + v + hs * 8;
                if (m < nvalid) {
                    const size_t rrow = ((size_t)s_tok[m] * 2 + s_slot[m]) * NQ + q;
                    contribq[rrow * HIDDEN + d0 + row] = acc[rt][v] * s_w[m];
                }
            }
        }
    }
}

// ------------------------------------------------------------- combine ----
__global__ void combine_kernel(const float* __restrict__ contribq,
                               float* __restrict__ out, int n) {
    const int i = blockIdx.x * blockDim.x + threadIdx.x;
    if (i >= n) return;
    const int t = i / HIDDEN;
    const int d = i % HIDDEN;
    const float* base = contribq + (size_t)t * 2 * NQ * HIDDEN + d;
    float s = 0.f;
    #pragma unroll
    for (int k = 0; k < 2 * NQ; ++k) s += base[(size_t)k * HIDDEN];
    out[i] = s;
}

// --------------------------------------------------------------- launch ----
extern "C" void kernel_launch(void* const* d_in, const int* in_sizes, int n_in,
                              void* d_out, int out_size, void* d_ws, size_t ws_size,
                              hipStream_t stream) {
    const float* x        = (const float*)d_in[0];
    const float* router_w = (const float*)d_in[1];
    const float* gate_w   = (const float*)d_in[2];
    const float* up_w     = (const float*)d_in[3];
    const float* down_w   = (const float*)d_in[4];
    // d_in[5] = top_k scalar, fixed at 2 here.

    const int T = in_sizes[0] / HIDDEN;     // 2048 tokens

    float* out        = (float*)d_out;                       // [T, HIDDEN]
    float* logits_out = out + (size_t)T * HIDDEN;            // [T, E]

    char*  ws       = (char*)d_ws;
    float* combine  = (float*)ws;       ws += ((size_t)T * N_EXPERTS * 4 + 255) & ~255ull;
    int*   counts   = (int*)ws;         ws += 256;
    int*   lists    = (int*)ws;         ws += ((size_t)N_EXPERTS * T * 4 + 255) & ~255ull;
    float* contribq = (float*)ws;       // [T*2, NQ, HIDDEN] = 64 MB

    zero_counts_kernel<<<1, 64, 0, stream>>>(counts);

    router_kernel<<<dim3((T + WAVES - 1) / WAVES), 32 * WAVES, 0, stream>>>(
        x, router_w, logits_out, combine, counts, lists, T);

    const int maxTiles = (T + M_TILE - 1) / M_TILE;
    const size_t shmem = (size_t)M_TILE * HIDDEN * 2 + (size_t)M_TILE * H_CHUNK * 2; // 256 KB
    moe_mlp_kernel<<<dim3(maxTiles, NQ, N_EXPERTS), 256, shmem, stream>>>(
        x, gate_w, up_w, down_w, combine, counts, lists, contribq, T);

    const int n = T * HIDDEN;
    combine_kernel<<<(n + 255) / 256, 256, 0, stream>>>(contribq, out, n);
}
